// Block_mamba_33852932227256
// MI455X (gfx1250) — matrix operand order repaired and checked by reading, not compiled, not run on hardware
//
#include <hip/hip_runtime.h>
#include <hip/hip_bf16.h>
#include <math.h>

// ---- problem constants (from reference) ----
#define BSZ     4
#define SEQ     2048
#define DIM     512
#define DSTATE  64
#define DCONV   4
#define DINNER  1024           // EXPAND*DIM
#define DTRANK  32             // (DIM+15)//16
#define NX      (DTRANK + 2*DSTATE)   // 160
#define MROWS   (BSZ*SEQ)      // 8192
#define GRP     8              // scan steps per pipeline group
#define NGRP    (SEQ/GRP)      // 256

typedef __bf16 v16bf __attribute__((ext_vector_type(16)));
typedef __bf16 v8bf  __attribute__((ext_vector_type(8)));
typedef float  v8f   __attribute__((ext_vector_type(8)));

// ---- WMMA fragment loaders (ISA 7.12.2, wave32, 16-bit operands) ----
// A (16x32, MxK): lane l -> row l&15, elems 0..7 = K[(l>>4)*8 .. +7],
//                 elems 8..15 = K[(l>>4)*8+16 .. +23]
__device__ __forceinline__ v16bf load_fragA(const __bf16* p) {
    union { v16bf v; v8bf h[2]; } u;
    u.h[0] = *(const v8bf*)p;
    u.h[1] = *(const v8bf*)(p + 16);
    return u.v;
}
// B (32x16, KxN): lane l -> col l&15, elems 0..7 = K[(l>>4)*8 .. +7], elems 8..15 = K[(l>>4)*8+16 .. +23]
__device__ __forceinline__ v16bf load_fragB(const __bf16* p) {
    union { v16bf v; v8bf h[2]; } u;
    u.h[0] = *(const v8bf*)p;
    u.h[1] = *(const v8bf*)(p + 16);
    return u.v;
}

// branchless numerically-stable softplus
__device__ __forceinline__ float softplus_f(float v) {
    return fmaxf(v, 0.0f) + __logf(1.0f + __expf(-fabsf(v)));
}

// ---------------------------------------------------------------------------
// 1) LayerNorm(x) -> bf16 A-matrix for in_proj GEMM.
// ---------------------------------------------------------------------------
__global__ __launch_bounds__(256) void ln_in_kernel(
    const float* __restrict__ x, const float* __restrict__ w,
    const float* __restrict__ b, __bf16* __restrict__ xn)
{
    __shared__ float red[256];
    const int r = blockIdx.x, tid = threadIdx.x;
    const float* xr = x + (size_t)r * DIM;
    float v0 = xr[tid], v1 = xr[tid + 256];

    red[tid] = v0 + v1;
    __syncthreads();
    for (int o = 128; o > 0; o >>= 1) { if (tid < o) red[tid] += red[tid + o]; __syncthreads(); }
    float mu = red[0] * (1.0f / DIM);
    __syncthreads();

    float d0 = v0 - mu, d1 = v1 - mu;
    red[tid] = d0 * d0 + d1 * d1;
    __syncthreads();
    for (int o = 128; o > 0; o >>= 1) { if (tid < o) red[tid] += red[tid + o]; __syncthreads(); }
    float rstd = rsqrtf(red[0] * (1.0f / DIM) + 1e-5f);

    __bf16* xo = xn + (size_t)r * DIM;
    xo[tid]       = (__bf16)(d0 * rstd * w[tid]       + b[tid]);
    xo[tid + 256] = (__bf16)(d1 * rstd * w[tid + 256] + b[tid + 256]);
}

// ---------------------------------------------------------------------------
// 2) Templated big-tile WMMA GEMM: 32 x (JT*16) C-tile per wave
//    (2*JT accumulators, 2 A-frags x JT B-frags per 32-wide k-step
//     => 2*JT v_wmma per (4 + 2*JT) b128 loads).
//    C[M,NN](f32) = A[M,KK](bf16) * W[NN,KK](bf16)^T.
//    NN/KK compile-time: all strides fold into instruction-immediate offsets.
//    HAS_BIAS / ACT(softplus) / DTR_OUT(dt_r bf16 slice) compile-time too.
// ---------------------------------------------------------------------------
template<int JT, int NN, int KK, bool HAS_BIAS, bool ACT, bool DTR_OUT>
__global__ __launch_bounds__(256) void gemm_wmma_t(
    const __bf16* __restrict__ A, const __bf16* __restrict__ W,
    const float* __restrict__ bias, float* __restrict__ C,
    __bf16* __restrict__ dtr, int M)
{
    const int wid  = (int)((blockIdx.x * (size_t)blockDim.x + threadIdx.x) >> 5);
    const int lane = threadIdx.x & 31;
    constexpr int ntn = NN / (JT * 16);
    const int tm   = wid / ntn;
    const int tn   = wid - tm * ntn;
    if (tm >= (M >> 5)) return;              // uniform per wave

    const int rr = lane & 15;
    const int ka = (lane >> 4) << 3;         // A k-offset
    const int kb = (lane >> 4) << 3;         // B k-offset
    const __bf16* ap = A + (size_t)(tm * 32 + rr) * KK + ka;
    const __bf16* wp = W + (size_t)(tn * JT * 16 + rr) * KK + kb;

    v8f acc[2][JT];
#pragma unroll
    for (int r = 0; r < 2; ++r)
#pragma unroll
        for (int j = 0; j < JT; ++j) acc[r][j] = (v8f){};

    for (int k = 0; k < KK; k += 32) {
        v16bf af[2], bf_[JT];
        af[0] = load_fragA(ap + k);
        af[1] = load_fragA(ap + 16 * KK + k);
#pragma unroll
        for (int j = 0; j < JT; ++j)
            bf_[j] = load_fragB(wp + j * 16 * KK + k);
#pragma unroll
        for (int r = 0; r < 2; ++r)
#pragma unroll
            for (int j = 0; j < JT; ++j)
                acc[r][j] = __builtin_amdgcn_wmma_f32_16x16x32_bf16(
                    false, af[r], false, bf_[j], (short)0, acc[r][j], false, false);
    }

    // C layout: vgpr i -> row (i + 8*(lane>=16)), col = lane&15
#pragma unroll
    for (int r = 0; r < 2; ++r) {
        const int rbase = tm * 32 + r * 16 + ((lane >> 4) << 3);
#pragma unroll
        for (int j = 0; j < JT; ++j) {
            const int col = tn * JT * 16 + j * 16 + (lane & 15);
            float* cp = C + (size_t)rbase * NN + col;
            float bv = 0.0f;
            if constexpr (HAS_BIAS) bv = bias[col];
#pragma unroll
            for (int i = 0; i < 8; ++i) {
                float v = acc[r][j][i];
                if constexpr (HAS_BIAS) v += bv;
                if constexpr (ACT) v = softplus_f(v);
                cp[i * NN] = v;
                if constexpr (DTR_OUT) {
                    if (col < DTRANK)
                        dtr[(size_t)(rbase + i) * DTRANK + col] = (__bf16)v;
                }
            }
        }
    }
}

// ---------------------------------------------------------------------------
// 3) Depthwise causal conv (width 4) + SiLU; writes f32 (scan) + bf16 (GEMM A)
// ---------------------------------------------------------------------------
__global__ __launch_bounds__(256) void conv_silu_kernel(
    const float* __restrict__ xz, const float* __restrict__ cw,
    const float* __restrict__ cb, float* __restrict__ xc,
    __bf16* __restrict__ xcbf)
{
    const size_t idx = blockIdx.x * (size_t)256 + threadIdx.x;  // over MROWS*DINNER
    const int    e   = (int)(idx & (DINNER - 1));
    const size_t row = idx >> 10;
    const int    l   = (int)(row & (SEQ - 1));
    const int    b   = (int)(row >> 11);

    float acc = cb[e];
#pragma unroll
    for (int j = 0; j < DCONV; ++j) {
        int lj = l - (DCONV - 1) + j;
        if (lj >= 0)
            acc += xz[((size_t)b * SEQ + lj) * (2 * DINNER) + e] * cw[e * DCONV + j];
    }
    float sv = acc / (1.0f + __expf(-acc));   // silu
    xc[idx]   = sv;
    xcbf[idx] = (__bf16)sv;
}

// generic f32 -> bf16 (weight repack)
__global__ __launch_bounds__(256) void f32_to_bf16_kernel(
    const float* __restrict__ s, __bf16* __restrict__ d, int n)
{
    int i = blockIdx.x * 256 + threadIdx.x;
    if (i < n) d[i] = (__bf16)s[i];
}

// ---------------------------------------------------------------------------
// 4) Selective scan, pipelined in groups of GRP=8 steps:
//    - B_t/C_t: 2-group x 8-slot LDS ring; wave0 issues next group's 8
//      global_load_async_to_lds_b128 before computing the current group, then
//      s_wait_asynccnt 8 (ASYNC loads complete in order -> current group is
//      resident while next group stays in flight).
//    - dt/xc/z: double-buffered register groups, next group's loads issued
//      before the current group's compute -> latency hidden.
//    2 states/lane, shfl_xor reduction; fuses y = ys + D*xc, y *= silu(z),
//    and the bf16 cast feeding out_proj.
// ---------------------------------------------------------------------------
__global__ __launch_bounds__(256) void scan_kernel(
    const float* __restrict__ x_dbl, const float* __restrict__ dt,
    const float* __restrict__ xc,    const float* __restrict__ xz,
    const float* __restrict__ A_log, const float* __restrict__ Dp,
    __bf16* __restrict__ ybf)
{
    __shared__ float sBC[2 * GRP * 2 * DSTATE];   // 2 groups x 8 steps x 128 f32 = 8KB
    const int b    = blockIdx.x >> 7;             // / (DINNER/8)
    const int eg   = blockIdx.x & 127;
    const int wave = threadIdx.x >> 5;
    const int lane = threadIdx.x & 31;
    const int e    = eg * 8 + wave;
    const int s0   = lane * 2, s1 = s0 + 1;

    const unsigned lds_base = (unsigned)(size_t)(&sBC[0]) + (unsigned)lane * 16u;
    const float*   gsrc0    = x_dbl + (size_t)b * SEQ * NX + DTRANK + lane * 4;
    const size_t   rc0      = (size_t)b * SEQ;    // first row of this batch

    const float a0 = -__expf(A_log[e * DSTATE + s0]);
    const float a1 = -__expf(A_log[e * DSTATE + s1]);
    const float De = Dp[e];
    float h0 = 0.0f, h1 = 0.0f;

    // ---- prologue: group 0 ----
    float dtc[GRP], xcc[GRP], zc[GRP];
#pragma unroll
    for (int i = 0; i < GRP; ++i) {
        const size_t row = rc0 + i;
        dtc[i] = dt[row * DINNER + e];
        xcc[i] = xc[row * DINNER + e];
        zc[i]  = xz[row * (2 * DINNER) + DINNER + e];
    }
    if (wave == 0) {
#pragma unroll
        for (int i = 0; i < GRP; ++i) {
            const float* gp = gsrc0 + (size_t)i * NX;
            unsigned loff = lds_base + (unsigned)i * 512u;   // buf0 slot i
            asm volatile("global_load_async_to_lds_b128 %0, %1, off"
                         :: "v"(loff), "v"(gp) : "memory");
        }
    }

    for (int g = 0; g < NGRP; ++g) {
        const int t0 = g * GRP;
        float dtn[GRP], xcn[GRP], zn[GRP];

        if (g + 1 < NGRP) {
            // issue next group's scalar-stream loads (hidden under compute)
#pragma unroll
            for (int i = 0; i < GRP; ++i) {
                const size_t row = rc0 + t0 + GRP + i;
                dtn[i] = dt[row * DINNER + e];
                xcn[i] = xc[row * DINNER + e];
                zn[i]  = xz[row * (2 * DINNER) + DINNER + e];
            }
            if (wave == 0) {
                // issue next group's async B/C copies into the other ring half
#pragma unroll
                for (int i = 0; i < GRP; ++i) {
                    const float* gp = gsrc0 + (size_t)(t0 + GRP + i) * NX;
                    unsigned loff = lds_base
                                  + (unsigned)(((g + 1) & 1) * GRP + i) * 512u;
                    asm volatile("global_load_async_to_lds_b128 %0, %1, off"
                                 :: "v"(loff), "v"(gp) : "memory");
                }
                asm volatile("s_wait_asynccnt 0x8" ::: "memory");
            }
        } else if (wave == 0) {
            asm volatile("s_wait_asynccnt 0x0" ::: "memory");
        }
        __syncthreads();

#pragma unroll
        for (int i = 0; i < GRP; ++i) {
            const float* bc = &sBC[(((unsigned)g & 1) * GRP + i) * (2 * DSTATE)];
            const float dtv = dtc[i], xv = xcc[i];
            const float du  = dtv * xv;
            h0 = __expf(dtv * a0) * h0 + du * bc[s0];
            h1 = __expf(dtv * a1) * h1 + du * bc[s1];

            float p = h0 * bc[DSTATE + s0] + h1 * bc[DSTATE + s1];
#pragma unroll
            for (int off = 16; off > 0; off >>= 1) p += __shfl_xor(p, off, 32);

            if (lane == 0) {
                float y = p + De * xv;
                float z = zc[i];
                y *= z / (1.0f + __expf(-z));      // * silu(z)
                ybf[(rc0 + t0 + i) * DINNER + e] = (__bf16)y;
            }
        }
        __syncthreads();

        if (g + 1 < NGRP) {
#pragma unroll
            for (int i = 0; i < GRP; ++i) { dtc[i] = dtn[i]; xcc[i] = xcn[i]; zc[i] = zn[i]; }
        }
    }
}

// ---------------------------------------------------------------------------
// 5) out = x + LayerNorm(m)  (ln1 weights)
// ---------------------------------------------------------------------------
__global__ __launch_bounds__(256) void ln_res_kernel(
    const float* __restrict__ x, const float* __restrict__ m,
    const float* __restrict__ w, const float* __restrict__ b,
    float* __restrict__ out)
{
    __shared__ float red[256];
    const int r = blockIdx.x, tid = threadIdx.x;
    const float* mr = m + (size_t)r * DIM;
    float v0 = mr[tid], v1 = mr[tid + 256];

    red[tid] = v0 + v1;
    __syncthreads();
    for (int o = 128; o > 0; o >>= 1) { if (tid < o) red[tid] += red[tid + o]; __syncthreads(); }
    float mu = red[0] * (1.0f / DIM);
    __syncthreads();

    float d0 = v0 - mu, d1 = v1 - mu;
    red[tid] = d0 * d0 + d1 * d1;
    __syncthreads();
    for (int o = 128; o > 0; o >>= 1) { if (tid < o) red[tid] += red[tid + o]; __syncthreads(); }
    float rstd = rsqrtf(red[0] * (1.0f / DIM) + 1e-5f);

    const float* xr = x + (size_t)r * DIM;
    float* orow = out + (size_t)r * DIM;
    orow[tid]       = xr[tid]       + d0 * rstd * w[tid]       + b[tid];
    orow[tid + 256] = xr[tid + 256] + d1 * rstd * w[tid + 256] + b[tid + 256];
}

// ---------------------------------------------------------------------------
extern "C" void kernel_launch(void* const* d_in, const int* in_sizes, int n_in,
                              void* d_out, int out_size, void* d_ws, size_t ws_size,
                              hipStream_t stream)
{
    (void)in_sizes; (void)n_in; (void)out_size; (void)ws_size;
    const float* x         = (const float*)d_in[0];
    const float* ln_m_w    = (const float*)d_in[1];
    const float* ln_m_b    = (const float*)d_in[2];
    const float* ln1_w     = (const float*)d_in[3];
    const float* ln1_b     = (const float*)d_in[4];
    const float* in_proj_w = (const float*)d_in[5];   // (2048, 512)
    const float* conv_w    = (const float*)d_in[6];   // (1024, 1, 4)
    const float* conv_b    = (const float*)d_in[7];
    const float* x_proj_w  = (const float*)d_in[8];   // (160, 1024)
    const float* dt_proj_w = (const float*)d_in[9];   // (1024, 32)
    const float* dt_proj_b = (const float*)d_in[10];
    const float* A_log     = (const float*)d_in[11];
    const float* D_param   = (const float*)d_in[12];
    const float* out_proj_w= (const float*)d_in[13];  // (512, 1024)
    float* out = (float*)d_out;

    // ---- carve workspace ----
    char* ws = (char*)d_ws;
    size_t off = 0;
    auto take = [&](size_t bytes) -> char* {
        char* p = ws + off;
        off += (bytes + 255) & ~(size_t)255;
        return p;
    };
    __bf16* xn_bf   = (__bf16*)take((size_t)MROWS * DIM * 2);
    __bf16* w_in_bf = (__bf16*)take((size_t)2 * DINNER * DIM * 2);
    __bf16* w_xp_bf = (__bf16*)take((size_t)NX * DINNER * 2);
    __bf16* w_dt_bf = (__bf16*)take((size_t)DINNER * DTRANK * 2);
    __bf16* w_out_bf= (__bf16*)take((size_t)DIM * DINNER * 2);
    float*  xz      = (float*) take((size_t)MROWS * 2 * DINNER * 4);
    float*  xc      = (float*) take((size_t)MROWS * DINNER * 4);
    __bf16* xc_bf   = (__bf16*)take((size_t)MROWS * DINNER * 2);
    float*  x_dbl   = (float*) take((size_t)MROWS * NX * 4);
    __bf16* dtr_bf  = (__bf16*)take((size_t)MROWS * DTRANK * 2);
    float*  dt_buf  = (float*) take((size_t)MROWS * DINNER * 4);
    __bf16* y_bf    = (__bf16*)take((size_t)MROWS * DINNER * 2);
    float*  m_buf   = (float*) take((size_t)MROWS * DIM * 4);

    // ---- weight repack to bf16 ----
    {
        int n = 2 * DINNER * DIM;
        f32_to_bf16_kernel<<<(n + 255) / 256, 256, 0, stream>>>(in_proj_w, w_in_bf, n);
        n = NX * DINNER;
        f32_to_bf16_kernel<<<(n + 255) / 256, 256, 0, stream>>>(x_proj_w, w_xp_bf, n);
        n = DINNER * DTRANK;
        f32_to_bf16_kernel<<<(n + 255) / 256, 256, 0, stream>>>(dt_proj_w, w_dt_bf, n);
        n = DIM * DINNER;
        f32_to_bf16_kernel<<<(n + 255) / 256, 256, 0, stream>>>(out_proj_w, w_out_bf, n);
    }

    // ---- 1. input layernorm -> bf16 ----
    ln_in_kernel<<<MROWS, 256, 0, stream>>>(x, ln_m_w, ln_m_b, xn_bf);

    // ---- 2. in_proj: xz = xn @ W_in^T   (8192 x 2048 x 512), 32x64 tiles ----
    {
        int waves = (MROWS / 32) * ((2 * DINNER) / 64);   // 8192
        gemm_wmma_t<4, 2 * DINNER, DIM, false, false, false>
            <<<waves / 8, 256, 0, stream>>>(
                xn_bf, w_in_bf, nullptr, xz, nullptr, MROWS);
    }

    // ---- 3. depthwise conv + silu ----
    conv_silu_kernel<<<(MROWS * DINNER) / 256, 256, 0, stream>>>(
        xz, conv_w, conv_b, xc, xc_bf);

    // ---- 4. x_proj: x_dbl = xc @ W_xp^T (8192 x 160 x 1024), 32x80 tiles,
    //         dt_r bf16 slice fused ----
    {
        int waves = (MROWS / 32) * (NX / 80);             // 512
        gemm_wmma_t<5, NX, DINNER, false, false, true>
            <<<waves / 8, 256, 0, stream>>>(
                xc_bf, w_xp_bf, nullptr, x_dbl, dtr_bf, MROWS);
    }

    // ---- 5. dt = softplus(dt_r @ W_dt^T + b)  (8192 x 1024 x 32) ----
    {
        int waves = (MROWS / 32) * (DINNER / 64);         // 4096
        gemm_wmma_t<4, DINNER, DTRANK, true, true, false>
            <<<waves / 8, 256, 0, stream>>>(
                dtr_bf, w_dt_bf, dt_proj_b, dt_buf, nullptr, MROWS);
    }

    // ---- 6. selective scan (pipelined async-to-LDS + prefetched streams) ----
    scan_kernel<<<BSZ * (DINNER / 8), 256, 0, stream>>>(
        x_dbl, dt_buf, xc, xz, A_log, D_param, y_bf);

    // ---- 7. out_proj: m = y @ W_out^T  (8192 x 512 x 1024) ----
    {
        int waves = (MROWS / 32) * (DIM / 64);            // 2048
        gemm_wmma_t<4, DIM, DINNER, false, false, false>
            <<<waves / 8, 256, 0, stream>>>(
                y_bf, w_out_bf, nullptr, m_buf, nullptr, MROWS);
    }

    // ---- 8. out = x + layernorm(m) ----
    ln_res_kernel<<<MROWS, 256, 0, stream>>>(x, m_buf, ln1_w, ln1_b, out);
}
